// SAM_678604833487
// MI455X (gfx1250) — matrix operand order repaired and checked by reading, not compiled, not run on hardware
//
#include <hip/hip_runtime.h>
#include <hip/hip_bf16.h>
#include <stdint.h>

// Problem constants (from reference)
#define C_DIM   256
#define QK_DIM  32
#define BATCH   4
#define NPOS    4096            // 64*64

typedef __bf16 bf16_t;
typedef __bf16 v8bf  __attribute__((ext_vector_type(8)));
typedef __bf16 v16bf __attribute__((ext_vector_type(16)));
typedef float  v8f   __attribute__((ext_vector_type(8)));
typedef unsigned int v4u __attribute__((ext_vector_type(4)));
typedef int    v8i   __attribute__((ext_vector_type(8)));
typedef int    v4i   __attribute__((ext_vector_type(4)));

// ---------------------------------------------------------------------------
// Kernel 1: 1x1-conv projections  Q,K -> [b][n][32] bf16 ; V -> [b][c][n] bf16
// grid (NPOS/256, BATCH, 5) block 256.  chunk 0 = Q+K, chunks 1..4 = V/64ch.
// ---------------------------------------------------------------------------
__global__ __launch_bounds__(256) void sa_proj_kernel(
    const float* __restrict__ x,
    const float* __restrict__ Wq, const float* __restrict__ bq,
    const float* __restrict__ Wk, const float* __restrict__ bk,
    const float* __restrict__ Wv, const float* __restrict__ bv,
    bf16_t* __restrict__ Qws, bf16_t* __restrict__ Kws, bf16_t* __restrict__ Vws)
{
    const int n  = blockIdx.x * 256 + threadIdx.x;
    const int b  = blockIdx.y;
    const int ch = blockIdx.z;                       // 0: Q+K ; 1..4: V chunk
    const float* xb = x + (size_t)b * C_DIM * NPOS + n;

    float acc[64];
#pragma unroll
    for (int i = 0; i < 64; ++i) acc[i] = 0.f;

    if (ch == 0) {
        for (int c = 0; c < C_DIM; ++c) {
            float xv = xb[(size_t)c * NPOS];         // coalesced along n
#pragma unroll
            for (int o = 0; o < 32; ++o) acc[o]      = fmaf(Wq[o * C_DIM + c], xv, acc[o]);
#pragma unroll
            for (int o = 0; o < 32; ++o) acc[32 + o] = fmaf(Wk[o * C_DIM + c], xv, acc[32 + o]);
        }
        bf16_t* qp = Qws + ((size_t)b * NPOS + n) * QK_DIM;
        bf16_t* kp = Kws + ((size_t)b * NPOS + n) * QK_DIM;
#pragma unroll
        for (int o = 0; o < 32; ++o) qp[o] = (bf16_t)(acc[o] + bq[o]);
#pragma unroll
        for (int o = 0; o < 32; ++o) kp[o] = (bf16_t)(acc[32 + o] + bk[o]);
    } else {
        const int c0 = (ch - 1) * 64;
        for (int c = 0; c < C_DIM; ++c) {
            float xv = xb[(size_t)c * NPOS];
#pragma unroll
            for (int o = 0; o < 64; ++o)
                acc[o] = fmaf(Wv[(size_t)(c0 + o) * C_DIM + c], xv, acc[o]);
        }
#pragma unroll
        for (int o = 0; o < 64; ++o)
            Vws[((size_t)b * C_DIM + (c0 + o)) * NPOS + n] = (bf16_t)(acc[o] + bv[c0 + o]);
    }
}

// ---------------------------------------------------------------------------
// TDM 2-D tile load: global (bf16 elements) -> LDS, per CDNA5 D# layout.
//   tensor element size = 2 bytes (data_size code 1)
//   This toolchain's builtin takes 6 args: (g0 v4u, g1 v8i, g2 v4i, g3 v4i,
//   extra v8i, cpol i32) — groups packed per ISA section 8, extra zeroed.
// ---------------------------------------------------------------------------
__device__ __forceinline__ void tdm_load_2d(unsigned lds_off,
                                            unsigned long long gaddr,
                                            unsigned td0, unsigned td1,
                                            unsigned long long stride0,
                                            unsigned tile0, unsigned tile1)
{
    v4u g0;
    g0[0] = 1u;                                            // count=1, user mode
    g0[1] = lds_off;                                       // lds_addr
    g0[2] = (unsigned)(gaddr & 0xffffffffull);             // global_addr[31:0]
    g0[3] = (unsigned)((gaddr >> 32) & 0x01ffffffull)      // global_addr[56:32]
            | 0x80000000u;                                 // type=2 ("image")
    v8i g1;
    g1[0] = (int)(1u << 16);                               // data_size=1 (2B), no mask/flags
    g1[1] = (int)((td0 & 0xffffu) << 16);                  // tensor_dim0[15:0] @63:48
    g1[2] = (int)((td0 >> 16) | ((td1 & 0xffffu) << 16));  // td0[31:16] | td1[15:0]
    g1[3] = (int)((td1 >> 16) | ((tile0 & 0xffffu) << 16));// td1[31:16] | tile_dim0
    g1[4] = (int)(tile1 & 0xffffu);                        // tile_dim1 ; tile_dim2=0
    g1[5] = (int)(unsigned)(stride0 & 0xffffffffull);      // tensor_dim0_stride[31:0]
    g1[6] = (int)(unsigned)((stride0 >> 32) & 0xffffull);  // stride[47:32] ; dim1_stride=0
    g1[7] = 0;
    v4i g2 = (v4i){0, 0, 0, 0};                            // 2-D tile: dims 2/3 unused
    v4i g3 = (v4i){0, 0, 0, 0};
    v8i gx = (v8i){0, 0, 0, 0, 0, 0, 0, 0};
    __builtin_amdgcn_tensor_load_to_lds(g0, g1, g2, g3, gx, 0);
}

// ---------------------------------------------------------------------------
// Kernel 2: flash attention, 4 waves/block, 16 queries/wave, 64 keys/iter.
//   Wave computes S^T = K(16x32) x Q^T(32x16)  (4 wmma)
//   then O^T += V^T(16x32keys) x P^T(32x16)    (32 wmma), online softmax.
// LDS double buffer: [2][ K tile 64x32 bf16 (4KB) | V tile 256x64 bf16 (32KB) ]
// ---------------------------------------------------------------------------
#define KT          64
#define NWAVES      4
#define KTILE_BYTES (KT * QK_DIM * 2)      // 4096
#define VTILE_BYTES (C_DIM * KT * 2)       // 32768
#define TILE_BYTES  (KTILE_BYTES + VTILE_BYTES)

__global__ __launch_bounds__(128) void sa_attn_kernel(
    const bf16_t* __restrict__ Qws, const bf16_t* __restrict__ Kws,
    const bf16_t* __restrict__ Vws,
    const float* __restrict__ x, const float* __restrict__ gamma,
    float* __restrict__ out)
{
    extern __shared__ char smem[];
    const int b    = blockIdx.y;
    const int wav  = threadIdx.x >> 5;
    const int lane = threadIdx.x & 31;
    const int col  = lane & 15;          // query column within wave tile
    const int hs   = lane >> 4;          // half-wave select (K-dim split)
    const int q0   = blockIdx.x * (NWAVES * 16) + wav * 16;

    // Q^T B-fragment (resident): lane=query column, halves = dims [hs*16 .. hs*16+15]
    const bf16_t* qp = Qws + ((size_t)b * NPOS + q0 + col) * QK_DIM + hs * 16;
    v8bf qlo = *(const v8bf*)(qp);
    v8bf qhi = *(const v8bf*)(qp + 8);
    v16bf qf = __builtin_shufflevector(qlo, qhi, 0,1,2,3,4,5,6,7,8,9,10,11,12,13,14,15);

    v8f O[16];                           // O^T accumulators: 16 channel tiles x (16x16 f32)
#pragma unroll
    for (int i = 0; i < 16; ++i) O[i] = (v8f){0.f,0.f,0.f,0.f,0.f,0.f,0.f,0.f};
    float m_run = -3.0e38f, l_run = 0.f;

    const unsigned long long kbase = (unsigned long long)(uintptr_t)(Kws + (size_t)b * NPOS * QK_DIM);
    const unsigned long long vbase = (unsigned long long)(uintptr_t)(Vws + (size_t)b * C_DIM * NPOS);
    const unsigned smem_off = (unsigned)(uintptr_t)smem;   // low 32 bits = LDS offset

    const int NIT = NPOS / KT;           // 64 iterations
    if (wav == 0) {                      // pre-issue two double-buffered tile pairs
#pragma unroll
        for (int pre = 0; pre < 2; ++pre) {
            unsigned boff = smem_off + pre * TILE_BYTES;
            tdm_load_2d(boff, kbase + (unsigned long long)pre * KT * QK_DIM * 2,
                        QK_DIM, NPOS, QK_DIM, QK_DIM, KT);
            tdm_load_2d(boff + KTILE_BYTES, vbase + (unsigned long long)pre * KT * 2,
                        NPOS, C_DIM, NPOS, KT, C_DIM);
        }
    }

    for (int it = 0; it < NIT; ++it) {
        const char* base = smem + (it & 1) * TILE_BYTES;
        const char* Kt = base;
        const char* Vt = base + KTILE_BYTES;

        if (wav == 0) {
            if (it < NIT - 1) __builtin_amdgcn_s_wait_tensorcnt(2);
            else              __builtin_amdgcn_s_wait_tensorcnt(0);
        }
        __syncthreads();                 // tiles for `it` visible to all waves

        // ---- S^T = K_tile x Q^T : 4 key sub-tiles of 16 ----
        v8f S[4];
#pragma unroll
        for (int t = 0; t < 4; ++t) {
            const char* row = Kt + (t * 16 + col) * (QK_DIM * 2);
            v8bf a0 = *(const v8bf*)(row + hs * 32);        // dims {0..7}/{16..23}
            v8bf a1 = *(const v8bf*)(row + hs * 32 + 16);   // dims {8..15}/{24..31}
            v16bf kf = __builtin_shufflevector(a0, a1, 0,1,2,3,4,5,6,7,8,9,10,11,12,13,14,15);
            v8f z = (v8f){0.f,0.f,0.f,0.f,0.f,0.f,0.f,0.f};
            S[t] = __builtin_amdgcn_wmma_f32_16x16x32_bf16(false, kf, false, qf,
                                                           (short)0, z, false, false);
        }

        // ---- online softmax (per-lane row = query col) ----
        float mx = -3.0e38f;
#pragma unroll
        for (int t = 0; t < 4; ++t)
#pragma unroll
            for (int r = 0; r < 8; ++r) mx = fmaxf(mx, S[t][r]);
        mx = fmaxf(mx, __shfl_xor(mx, 16));
        float m_new = fmaxf(m_run, mx);
        float scale = __expf(m_run - m_new);

        float p[4][8];
        float rs = 0.f;
#pragma unroll
        for (int t = 0; t < 4; ++t)
#pragma unroll
            for (int r = 0; r < 8; ++r) {
                p[t][r] = __expf(S[t][r] - m_new);
                rs += p[t][r];
            }
        rs += __shfl_xor(rs, 16);
        l_run = l_run * scale + rs;
        m_run = m_new;

#pragma unroll
        for (int ct = 0; ct < 16; ++ct)
#pragma unroll
            for (int r = 0; r < 8; ++r) O[ct][r] *= scale;

        // partner half-wave values (same query column, other key range)
        float pex[4][8];
#pragma unroll
        for (int t = 0; t < 4; ++t)
#pragma unroll
            for (int r = 0; r < 8; ++r) pex[t][r] = __shfl_xor(p[t][r], 16);

        // ---- P^T B-fragments: halves = keys kc*32 + [hs*16 .. hs*16+15] ----
        v16bf pf[2];
#pragma unroll
        for (int kc = 0; kc < 2; ++kc) {
            v16bf f;
#pragma unroll
            for (int r = 0; r < 8; ++r) {
                float own  = hs ? p[2*kc+1][r]   : p[2*kc][r];
                float part = hs ? pex[2*kc+1][r] : pex[2*kc][r];
                f[r]     = (bf16_t)(hs ? part : own);
                f[8 + r] = (bf16_t)(hs ? own  : part);
            }
            pf[kc] = f;
        }

        // ---- O^T += V^T x P^T : 16 channel tiles x 2 key chunks ----
#pragma unroll
        for (int ct = 0; ct < 16; ++ct) {
            const char* vrow = Vt + (ct * 16 + col) * (KT * 2);
#pragma unroll
            for (int kc = 0; kc < 2; ++kc) {
                v8bf a0 = *(const v8bf*)(vrow + kc * 64 + hs * 32);
                v8bf a1 = *(const v8bf*)(vrow + kc * 64 + hs * 32 + 16);
                v16bf vf = __builtin_shufflevector(a0, a1, 0,1,2,3,4,5,6,7,8,9,10,11,12,13,14,15);
                O[ct] = __builtin_amdgcn_wmma_f32_16x16x32_bf16(false, vf, false, pf[kc],
                                                                (short)0, O[ct], false, false);
            }
        }

        __syncthreads();                 // everyone done with this buffer
        if (wav == 0 && it + 2 < NIT) {
            const int nx = it + 2;
            unsigned boff = smem_off + (nx & 1) * TILE_BYTES;
            tdm_load_2d(boff, kbase + (unsigned long long)nx * KT * QK_DIM * 2,
                        QK_DIM, NPOS, QK_DIM, QK_DIM, KT);
            tdm_load_2d(boff + KTILE_BYTES, vbase + (unsigned long long)nx * KT * 2,
                        NPOS, C_DIM, NPOS, KT, C_DIM);
        }
    }

    // ---- epilogue: out[b][c][m] = gamma * O/l + x ----
    const float g = gamma[0];
    const float inv_l = (l_run > 0.f) ? (1.0f / l_run) : 0.f;
    const int m = q0 + col;
#pragma unroll
    for (int ct = 0; ct < 16; ++ct) {
#pragma unroll
        for (int r = 0; r < 8; ++r) {
            int c = ct * 16 + r + 8 * hs;                  // C-fragment row mapping
            size_t idx = ((size_t)b * C_DIM + c) * NPOS + m;
            out[idx] = g * (O[ct][r] * inv_l) + x[idx];
        }
    }
}

// ---------------------------------------------------------------------------
extern "C" void kernel_launch(void* const* d_in, const int* in_sizes, int n_in,
                              void* d_out, int out_size, void* d_ws, size_t ws_size,
                              hipStream_t stream)
{
    const float* x     = (const float*)d_in[0];
    const float* Wq    = (const float*)d_in[1];
    const float* bq    = (const float*)d_in[2];
    const float* Wk    = (const float*)d_in[3];
    const float* bk    = (const float*)d_in[4];
    const float* Wv    = (const float*)d_in[5];
    const float* bv    = (const float*)d_in[6];
    const float* gamma = (const float*)d_in[7];

    bf16_t* Qws = (bf16_t*)d_ws;
    bf16_t* Kws = Qws + (size_t)BATCH * NPOS * QK_DIM;
    bf16_t* Vws = Kws + (size_t)BATCH * NPOS * QK_DIM;     // ~10.5 MB total

    dim3 pg(NPOS / 256, BATCH, 5);
    hipLaunchKernelGGL(sa_proj_kernel, pg, dim3(256), 0, stream,
                       x, Wq, bq, Wk, bk, Wv, bv, Qws, Kws, Vws);

    dim3 ag(NPOS / (NWAVES * 16), BATCH);                  // 64 x 4 = 256 blocks
    hipLaunchKernelGGL(sa_attn_kernel, ag, dim3(128), 2 * TILE_BYTES, stream,
                       Qws, Kws, Vws, x, gamma, (float*)d_out);
}
